// TGNPLMemory_63840393888431
// MI455X (gfx1250) — matrix operand order repaired and checked by reading, not compiled, not run on hardware
//
#include <hip/hip_runtime.h>
#include <hip/hip_bf16.h>

#define DEVFN __device__ __forceinline__

constexpr int N_NODES   = 100000;
constexpr int NUM_PRODS = 512;
constexpr int NUM_FIRMS = N_NODES - NUM_PRODS;   // 99488
constexpr int STATE     = 128;
constexpr int RAW       = 64;
constexpr int TIME      = 32;
constexpr int E_EDGES   = 50000;
constexpr int MSG       = 3 * STATE + RAW + TIME;  // 480
constexpr int GATES     = 3 * STATE;               // 384
constexpr int MEMDIM    = STATE + NUM_PRODS;       // 640
constexpr float DEBT    = 5.0f;
constexpr float REWARD  = 1.0f;

typedef _Float16 v16h __attribute__((ext_vector_type(16)));
typedef float    v8f  __attribute__((ext_vector_type(8)));

// ---------------------------------------------------------------------------
// WMMA fragment helpers (CDNA5 16x16x32 f16 layouts, wave32)
// A 16x32: lane L holds row m=L&15; 16 contiguous halfs at K = kb..kb+15,
//          kb = 16*(L>>4). Two aligned 16B chunks.
// B 32x16: lane L holds col n=L&15; 16 contiguous halfs at K = 16*(L>>4).
// C/D f32: lane L col n=L&15, vgpr v -> row m = v + 8*(L>>4).
// ---------------------------------------------------------------------------
union FragU { uint4 u[2]; v16h h; };

DEVFN v16h lds_a_frag(const _Float16* row, int k0, int kb) {
    FragU f;
    f.u[0] = *(const uint4*)(row + k0 + kb);
    f.u[1] = *(const uint4*)(row + k0 + kb + 8);
    return f.h;
}

DEVFN v16h glb_b_frag(const _Float16* p) {  // 16 contiguous halfs, 32B aligned
    FragU f;
    f.u[0] = *(const uint4*)(p);
    f.u[1] = *(const uint4*)(p + 8);
    return f.h;
}

DEVFN v8f wmma_f16(v16h a, v16h b, v8f c) {
    return __builtin_amdgcn_wmma_f32_16x16x32_f16(
        false, a, false, b, (short)0, c, false, false);
}

DEVFN float sigmoidf_(float x) { return 1.0f / (1.0f + __expf(-x)); }

// ---------------------------------------------------------------------------
// Convert GRU weights to f16 once (reused by all 6250 node tiles)
// ---------------------------------------------------------------------------
__global__ void cvt_weights_kernel(const float* __restrict__ W_ih,
                                   const float* __restrict__ W_hh,
                                   _Float16* __restrict__ Wih_h,
                                   _Float16* __restrict__ Whh_h) {
    int i = blockIdx.x * blockDim.x + threadIdx.x;
    if (i < GATES * MSG)   Wih_h[i] = (_Float16)W_ih[i];
    if (i < GATES * STATE) Whh_h[i] = (_Float16)W_hh[i];
}

// ---------------------------------------------------------------------------
// att = relu(prod_emb @ (PB @ prod_emb^T))  -- tiny (67 MFLOP), VALU is fine
// ---------------------------------------------------------------------------
__global__ void att_tmp_kernel(const float* __restrict__ memory,
                               const float* __restrict__ PB,
                               float* __restrict__ tmp) {
    int idx = blockIdx.x * blockDim.x + threadIdx.x;   // 128*512
    if (idx >= STATE * NUM_PRODS) return;
    int i = idx >> 9, j = idx & 511;
    const float* emb = memory + (size_t)(NUM_FIRMS + j) * MEMDIM;
    float s = 0.f;
    #pragma unroll 8
    for (int k = 0; k < STATE; ++k) s += PB[i * STATE + k] * emb[k];
    tmp[idx] = s;
}

__global__ void att_kernel(const float* __restrict__ memory,
                           const float* __restrict__ tmp,
                           float* __restrict__ att) {
    int idx = blockIdx.x * blockDim.x + threadIdx.x;   // 512*512
    if (idx >= NUM_PRODS * NUM_PRODS) return;
    int a = idx >> 9, b = idx & 511;
    const float* emb = memory + (size_t)(NUM_FIRMS + a) * MEMDIM;
    float s = 0.f;
    #pragma unroll 8
    for (int k = 0; k < STATE; ++k) s += emb[k] * tmp[k * NUM_PRODS + b];
    att[idx] = fmaxf(s, 0.f);
}

// ---------------------------------------------------------------------------
// Edge scatter: segment-sum of 3x480 messages, counts, segment-max(t),
// and the SPARSE formulation of consumed = supplied @ att
// ---------------------------------------------------------------------------
__global__ void edge_scatter_kernel(const float* __restrict__ memory,
                                    const float* __restrict__ raw_msg,
                                    const float* __restrict__ W_time,
                                    const float* __restrict__ b_time,
                                    const int* __restrict__ src,
                                    const int* __restrict__ dst,
                                    const int* __restrict__ prod,
                                    const int* __restrict__ t,
                                    const int* __restrict__ last_update,
                                    const float* __restrict__ att,
                                    float* __restrict__ seg,
                                    float* __restrict__ cnt,
                                    int* __restrict__ lu_max,
                                    float* __restrict__ consumed) {
    const int e = blockIdx.x;
    const int tid = threadIdx.x;      // 128 threads
    const int is = src[e], id = dst[e], ip = prod[e];
    const int te = t[e];
    const float tf = (float)te;

    __shared__ float base[3 * STATE + RAW];   // 448 shared across 3 targets
    for (int j = tid; j < 448; j += 128) {
        float v;
        if (j < 128)      v = memory[(size_t)is * MEMDIM + j];
        else if (j < 256) v = memory[(size_t)id * MEMDIM + (j - 128)];
        else if (j < 384) v = memory[(size_t)ip * MEMDIM + (j - 256)];
        else              v = raw_msg[(size_t)e * RAW + (j - 384)];
        base[j] = v;
    }
    __syncthreads();

    const int tgt[3] = {is, id, ip};
    #pragma unroll
    for (int c = 0; c < 3; ++c) {
        const int X = tgt[c];
        const float dt = tf - (float)last_update[X];
        for (int j = tid; j < MSG; j += 128) {
            float v = (j < 448) ? base[j]
                                : __cosf(dt * W_time[j - 448] + b_time[j - 448]);
            atomicAdd(&seg[(size_t)X * MSG + j], v);
        }
        if (tid == 0) {
            atomicAdd(&cnt[X], 1.0f);
            atomicMax(&lu_max[X], te);
        }
    }

    // consumed[src] += amt * att[p,:]   (sparse supplied@att)
    const float amt = fmaxf(raw_msg[(size_t)e * RAW], 1.0f);
    const int p = ip - NUM_FIRMS;
    for (int j = tid; j < NUM_PRODS; j += 128)
        atomicAdd(&consumed[(size_t)is * NUM_PRODS + j],
                  amt * att[p * NUM_PRODS + j]);
}

// ---------------------------------------------------------------------------
// Fused GRU update: aggr = seg/max(cnt,1) -> f16 LDS panel; each of 4 waves
// owns n-groups {w, w+8, w+16} so r/z/g gate triples land in the SAME lanes
// and the full GRU nonlinearity is evaluated in registers.
// 114 v_wmma_f32_16x16x32_f16 per wave.  M-tile = 16 nodes (N%16==0).
// ---------------------------------------------------------------------------
__global__ void __launch_bounds__(128)
gru_kernel(const float* __restrict__ memory,
           const float* __restrict__ seg,
           const float* __restrict__ cnt,
           const int*   __restrict__ lu_max,
           const _Float16* __restrict__ Wih_h,   // [384][480]
           const _Float16* __restrict__ Whh_h,   // [384][128]
           const float* __restrict__ b_ih,
           const float* __restrict__ b_hh,
           float* __restrict__ out) {            // N*640 state+inv, then N lu
    __shared__ __align__(16) _Float16 aggrh[16][MSG];    // 15.4 KB
    __shared__ __align__(16) _Float16 stateh[16][STATE]; //  4.0 KB
    __shared__ float invc[16];

    const int tid = threadIdx.x;
    const int m0  = blockIdx.x * 16;

    if (tid < 16) {
        const int node = m0 + tid;
        const float c = cnt[node];
        invc[tid] = 1.0f / fmaxf(c, 1.0f);
        out[(size_t)N_NODES * MEMDIM + node] =
            (c > 0.f) ? (float)lu_max[node] : 0.f;       // new_last_update
    }
    __syncthreads();

    for (int i = tid; i < 16 * MSG; i += 128) {
        const int r = i / MSG, k = i - r * MSG;
        aggrh[r][k] = (_Float16)(seg[(size_t)(m0 + r) * MSG + k] * invc[r]);
    }
    for (int i = tid; i < 16 * STATE; i += 128) {
        const int r = i >> 7, k = i & 127;
        stateh[r][k] = (_Float16)memory[(size_t)(m0 + r) * MEMDIM + k];
    }
    __syncthreads();

    const int w    = tid >> 5;
    const int lane = tid & 31;
    const int mrow = lane & 15;
    const int kb   = (lane >> 4) * 16;   // A fragment chunk base
    const int bn   = lane & 15;          // B column within tile
    const int kb2  = (lane >> 4) * 16;   // B fragment K base

    #pragma unroll
    for (int grp = 0; grp < 2; ++grp) {
        const int T  = w + grp * 4;      // n-group 0..7 -> n in [16T,16T+16)
        const int n0 = T * 16;
        const int nr = n0 + bn;          // W row for r-gate (z: +128, g: +256)

        v8f aR = {}, aZ = {}, aG = {}, hR = {}, hZ = {}, hG = {};

        // gi = aggr @ W_ih^T : K = 480 (15 steps)
        #pragma unroll
        for (int k0 = 0; k0 < MSG; k0 += 32) {
            const v16h A = lds_a_frag(&aggrh[mrow][0], k0, kb);
            aR = wmma_f16(A, glb_b_frag(Wih_h + (size_t)(nr)        * MSG + k0 + kb2), aR);
            aZ = wmma_f16(A, glb_b_frag(Wih_h + (size_t)(nr + 128)  * MSG + k0 + kb2), aZ);
            aG = wmma_f16(A, glb_b_frag(Wih_h + (size_t)(nr + 256)  * MSG + k0 + kb2), aG);
        }
        // gh = state @ W_hh^T : K = 128 (4 steps)
        #pragma unroll
        for (int k0 = 0; k0 < STATE; k0 += 32) {
            const v16h A = lds_a_frag(&stateh[mrow][0], k0, kb);
            hR = wmma_f16(A, glb_b_frag(Whh_h + (size_t)(nr)        * STATE + k0 + kb2), hR);
            hZ = wmma_f16(A, glb_b_frag(Whh_h + (size_t)(nr + 128)  * STATE + k0 + kb2), hZ);
            hG = wmma_f16(A, glb_b_frag(Whh_h + (size_t)(nr + 256)  * STATE + k0 + kb2), hG);
        }

        // GRU combine in registers: every accumulator shares the (m,n) map
        const float bir = b_ih[nr], biz = b_ih[nr + 128], big = b_ih[nr + 256];
        const float bhr = b_hh[nr], bhz = b_hh[nr + 128], bhg = b_hh[nr + 256];
        #pragma unroll
        for (int v = 0; v < 8; ++v) {
            const int m = v + 8 * (lane >> 4);
            const int node = m0 + m;
            const float r = sigmoidf_((aR[v] + bir) + (hR[v] + bhr));
            const float z = sigmoidf_((aZ[v] + biz) + (hZ[v] + bhz));
            const float g = tanhf((aG[v] + big) + r * (hG[v] + bhg));
            const float s = memory[(size_t)node * MEMDIM + nr];
            out[(size_t)node * MEMDIM + nr] = (1.0f - z) * g + z * s;
        }
    }
}

// ---------------------------------------------------------------------------
// Inventory finalize + loss reduction: new_inv = inv - consumed (bought added
// later, loss is computed BEFORE bought, matching the reference)
// ---------------------------------------------------------------------------
__global__ void fin_kernel(const float* __restrict__ memory,
                           const float* __restrict__ consumed,
                           float* __restrict__ out,
                           float* __restrict__ loss) {
    __shared__ float s1[256], s2[256];
    const int tid = threadIdx.x;
    float p1 = 0.f, p2 = 0.f;
    const size_t total = (size_t)N_NODES * NUM_PRODS;
    for (size_t i = (size_t)blockIdx.x * blockDim.x + tid; i < total;
         i += (size_t)gridDim.x * blockDim.x) {
        const size_t node = i >> 9;
        const int j = (int)(i & 511);
        const float inv = memory[node * MEMDIM + STATE + j];
        const float c = consumed[i];
        out[node * MEMDIM + STATE + j] = inv - c;
        p1 += fmaxf(c - inv, 0.f);
        p2 += fminf(c, inv);
    }
    s1[tid] = p1; s2[tid] = p2;
    __syncthreads();
    for (int s = 128; s > 0; s >>= 1) {
        if (tid < s) { s1[tid] += s1[tid + s]; s2[tid] += s2[tid + s]; }
        __syncthreads();
    }
    if (tid == 0) { atomicAdd(&loss[0], s1[0]); atomicAdd(&loss[1], s2[0]); }
}

__global__ void bought_kernel(const float* __restrict__ raw_msg,
                              const int* __restrict__ dst,
                              const int* __restrict__ prod,
                              float* __restrict__ out) {
    const int e = blockIdx.x * blockDim.x + threadIdx.x;
    if (e >= E_EDGES) return;
    const int p = prod[e] - NUM_FIRMS;
    const float amt = fmaxf(raw_msg[(size_t)e * RAW], 1.0f);
    atomicAdd(&out[(size_t)dst[e] * MEMDIM + STATE + p], amt);
}

__global__ void loss_final_kernel(const float* __restrict__ loss,
                                  float* __restrict__ out) {
    const float invM = 1.0f / ((float)N_NODES * (float)NUM_PRODS);
    out[(size_t)N_NODES * MEMDIM + N_NODES] =
        DEBT * loss[0] * invM - REWARD * loss[1] * invM;
}

// ---------------------------------------------------------------------------
extern "C" void kernel_launch(void* const* d_in, const int* in_sizes, int n_in,
                              void* d_out, int out_size, void* d_ws, size_t ws_size,
                              hipStream_t stream) {
    const float* memory   = (const float*)d_in[0];
    const float* raw_msg  = (const float*)d_in[1];
    const float* W_time   = (const float*)d_in[2];
    const float* b_time   = (const float*)d_in[3];
    const float* W_ih     = (const float*)d_in[4];
    const float* W_hh     = (const float*)d_in[5];
    const float* b_ih     = (const float*)d_in[6];
    const float* b_hh     = (const float*)d_in[7];
    const float* PB       = (const float*)d_in[8];
    const int*   src      = (const int*)d_in[9];
    const int*   dst      = (const int*)d_in[10];
    const int*   prod     = (const int*)d_in[11];
    const int*   t        = (const int*)d_in[12];
    const int*   last_upd = (const int*)d_in[13];
    float* out = (float*)d_out;

    char* ws = (char*)d_ws;
    size_t off = 0;
    auto take = [&](size_t bytes) -> void* {
        void* p = ws + off;
        off = (off + bytes + 255) & ~(size_t)255;
        return p;
    };
    float*    seg      = (float*)take((size_t)N_NODES * MSG * 4);
    float*    consumed = (float*)take((size_t)N_NODES * NUM_PRODS * 4);
    float*    cnt      = (float*)take((size_t)N_NODES * 4);
    int*      lu_max   = (int*)  take((size_t)N_NODES * 4);
    float*    att      = (float*)take((size_t)NUM_PRODS * NUM_PRODS * 4);
    float*    tmp      = (float*)take((size_t)STATE * NUM_PRODS * 4);
    _Float16* Wih_h    = (_Float16*)take((size_t)GATES * MSG * 2);
    _Float16* Whh_h    = (_Float16*)take((size_t)GATES * STATE * 2);
    float*    loss     = (float*)take(2 * 4);

    hipMemsetAsync(seg,      0, (size_t)N_NODES * MSG * 4, stream);
    hipMemsetAsync(consumed, 0, (size_t)N_NODES * NUM_PRODS * 4, stream);
    hipMemsetAsync(cnt,      0, (size_t)N_NODES * 4, stream);
    hipMemsetAsync(lu_max,   0, (size_t)N_NODES * 4, stream);   // t >= 0
    hipMemsetAsync(loss,     0, 2 * 4, stream);

    cvt_weights_kernel<<<(GATES * MSG + 255) / 256, 256, 0, stream>>>(
        W_ih, W_hh, Wih_h, Whh_h);
    att_tmp_kernel<<<(STATE * NUM_PRODS + 255) / 256, 256, 0, stream>>>(
        memory, PB, tmp);
    att_kernel<<<(NUM_PRODS * NUM_PRODS + 255) / 256, 256, 0, stream>>>(
        memory, tmp, att);
    edge_scatter_kernel<<<E_EDGES, 128, 0, stream>>>(
        memory, raw_msg, W_time, b_time, src, dst, prod, t, last_upd,
        att, seg, cnt, lu_max, consumed);
    gru_kernel<<<N_NODES / 16, 128, 0, stream>>>(
        memory, seg, cnt, lu_max, Wih_h, Whh_h, b_ih, b_hh, out);
    fin_kernel<<<4096, 256, 0, stream>>>(memory, consumed, out, loss);
    bought_kernel<<<(E_EDGES + 255) / 256, 256, 0, stream>>>(
        raw_msg, dst, prod, out);
    loss_final_kernel<<<1, 1, 0, stream>>>(loss, out);
}
